// RATransformer_29111288332521
// MI455X (gfx1250) — matrix operand order repaired and checked by reading, not compiled, not run on hardware
//
#include <hip/hip_runtime.h>
#include <math.h>

// ---------------- model constants (from reference) ----------------
#define BATCH   2
#define MDIM    32          // "m" axis (assets)
#define TLEN    64          // sequence length
#define NTOK    (BATCH*MDIM*TLEN)   // 4096 token rows
#define DM      512
#define NHEADS  8
#define DK      64
#define CTX     10
#define SLOT    ((size_t)NTOK * DM) // floats per activation slot (8 MB)

typedef __bf16 bf16_t;
typedef __attribute__((ext_vector_type(16))) __bf16 v16bf;
typedef __attribute__((ext_vector_type(8)))  __bf16 v8bf;
typedef __attribute__((ext_vector_type(8)))  float  v8f;

__device__ __forceinline__ bf16_t f2bf(float f) {
  union { float f; unsigned u; } v; v.f = f;
  unsigned r = v.u + 0x7FFFu + ((v.u >> 16) & 1u);   // round-to-nearest-even
  unsigned short h = (unsigned short)(r >> 16);
  return __builtin_bit_cast(bf16_t, h);
}

// ---------------- f32 -> bf16 conversion (row-major copy) ----------------
__global__ void ra_cvt_bf16(const float* __restrict__ src,
                            bf16_t* __restrict__ dst, int n)
{
  int i = blockIdx.x * blockDim.x + threadIdx.x;
  if (i < n) dst[i] = f2bf(src[i]);
}

// ---------------- f32 [K,O] -> bf16 transposed [O,K] ----------------
__global__ void ra_cvt_bf16_T(const float* __restrict__ W,
                              bf16_t* __restrict__ WT, int K, int O)
{
  int i = blockIdx.x * blockDim.x + threadIdx.x;
  if (i < K * O) {
    int k = i / O, o = i - k * O;            // coalesced read of W
    WT[(size_t)o * K + k] = f2bf(W[i]);      // strided write of WT
  }
}

// ---------------------------------------------------------------------------
// WMMA GEMM:  Y[row,col] = act( X[row,:K] @ W[:K,col] + bias[col] ) (+ res)
// Xbf row-major [NTOK,K] bf16; WT [O,K] bf16 (transposed weights).
// K%32==0, O%64==0. Block = 4 waves; each wave computes a 32x64 C tile
// (8 accumulators, 8 v_wmma per K=32 step, 12 b128 loads).
// ---------------------------------------------------------------------------
__global__ __launch_bounds__(128) void ra_gemm_wmma(
    const bf16_t* __restrict__ Xbf, const bf16_t* __restrict__ WT,
    const float* __restrict__ bias, const float* __restrict__ res,
    float* __restrict__ Y, int K, int O, int relu)
{
  const int wave = threadIdx.x >> 5;
  const int lane = threadIdx.x & 31;
  const int row0 = blockIdx.y * 128 + wave * 32;
  const int col0 = blockIdx.x * 64;
  const int m    = lane & 15;
  const int h    = lane >> 4;

  v8f acc[2][4] = {};

  // A fragment (16-bit A 16x32 layout): lane half h needs K runs
  // [kk+8h, +8) and [kk+16+8h, +8)  -> two b128 loads per fragment.
  const bf16_t* arow0 = Xbf + (size_t)(row0 + m) * K + 8 * h;
  const bf16_t* arow1 = arow0 + (size_t)16 * K;
  // B fragment (dense 32x16 B layout): lane half h needs K runs
  // [kk+8h, +8) and [kk+16+8h, +8) of column col -> two b128 loads per fragment.
  const bf16_t* bptr[4];
#pragma unroll
  for (int j = 0; j < 4; ++j)
    bptr[j] = WT + (size_t)(col0 + 16 * j + m) * K + 8 * h;

  for (int kk = 0; kk < K; kk += 32) {
    if (kk + 32 < K) {
      __builtin_prefetch(arow0 + kk + 32, 0, 1);   // global_prefetch_b8
      __builtin_prefetch(bptr[0] + kk + 32, 0, 1);
    }
    v8bf a0l = *(const v8bf*)(arow0 + kk);
    v8bf a0h = *(const v8bf*)(arow0 + kk + 16);
    v8bf a1l = *(const v8bf*)(arow1 + kk);
    v8bf a1h = *(const v8bf*)(arow1 + kk + 16);
    v16bf A0 = __builtin_shufflevector(a0l, a0h,
        0,1,2,3,4,5,6,7,8,9,10,11,12,13,14,15);
    v16bf A1 = __builtin_shufflevector(a1l, a1h,
        0,1,2,3,4,5,6,7,8,9,10,11,12,13,14,15);
#pragma unroll
    for (int j = 0; j < 4; ++j) {
      v8bf b0 = *(const v8bf*)(bptr[j] + kk);
      v8bf b1 = *(const v8bf*)(bptr[j] + kk + 16);
      v16bf B = __builtin_shufflevector(b0, b1,
          0,1,2,3,4,5,6,7,8,9,10,11,12,13,14,15);
      acc[0][j] = __builtin_amdgcn_wmma_f32_16x16x32_bf16(
          false, A0, false, B, (short)0, acc[0][j], false, false);
      acc[1][j] = __builtin_amdgcn_wmma_f32_16x16x32_bf16(
          false, A1, false, B, (short)0, acc[1][j], false, false);
    }
  }

  // C/D layout: VGPR r -> M = r (lanes0-15) / r+8 (lanes16-31); N = lane&15
#pragma unroll
  for (int a = 0; a < 2; ++a) {
#pragma unroll
    for (int j = 0; j < 4; ++j) {
      const int col = col0 + j * 16 + m;
      const float bv = bias[col];
#pragma unroll
      for (int r = 0; r < 8; ++r) {
        int row = row0 + 16 * a + r + (h << 3);
        float v = acc[a][j][r] + bv;
        if (relu) v = fmaxf(v, 0.f);
        if (res)  v += res[(size_t)row * O + col];
        Y[(size_t)row * O + col] = v;
      }
    }
  }
}

// ---------------- sinusoidal positional encoding add ----------------
__global__ void ra_add_pe(float* __restrict__ X)
{
  int i = blockIdx.x * blockDim.x + threadIdx.x;
  if (i >= (int)(NTOK * DM)) return;
  int c = i & (DM - 1);
  int t = (i >> 9) & (TLEN - 1);
  int ci = c & ~1;
  float div = expf(-(float)ci * (9.210340371976184f / (float)DM)); // ln(10000)
  float ang = (float)t * div;
  X[i] += (c & 1) ? cosf(ang) : sinf(ang);
}

// ---------------- LayerNorm: one wave per 512-row ----------------
__global__ __launch_bounds__(32) void ra_layernorm(
    const float* __restrict__ X, const float* __restrict__ g,
    const float* __restrict__ b, float* __restrict__ Y)
{
  int row  = blockIdx.x;
  int lane = threadIdx.x;
  const float* x = X + (size_t)row * DM;
  float xs[16], s = 0.f, s2 = 0.f;
#pragma unroll
  for (int e = 0; e < 16; ++e) {
    float v = x[lane + 32 * e];
    xs[e] = v; s += v; s2 += v * v;
  }
#pragma unroll
  for (int off = 16; off; off >>= 1) {
    s  += __shfl_xor(s,  off, 32);
    s2 += __shfl_xor(s2, off, 32);
  }
  float mu  = s * (1.f / DM);
  float var = s2 * (1.f / DM) - mu * mu;
  float inv = rsqrtf(var + 1e-6f);
  float* y = Y + (size_t)row * DM;
#pragma unroll
  for (int e = 0; e < 16; ++e) {
    int c = lane + 32 * e;
    y[c] = g[c] * (xs[e] - mu) * inv + b[c];
  }
}

// ---------------- context attention (window L=10, zero padded) ----------------
__global__ __launch_bounds__(32) void ra_ctx_attn(
    const float* __restrict__ Xin, float* __restrict__ Out)
{
  int token = blockIdx.x;            // (b*m)*64 + t
  int t     = token & (TLEN - 1);
  int lane  = threadIdx.x;
  const float* base = Xin + (size_t)(token - t) * DM;
  const float* xt   = Xin + (size_t)token * DM;

  float xreg[16];
#pragma unroll
  for (int e = 0; e < 16; ++e) xreg[e] = xt[lane + 32 * e];

  float s[CTX];
  for (int l = 0; l < CTX; ++l) {
    int idx = t - (CTX - 1) + l;
    float d = 0.f;
    if (idx >= 0) {
      const float* xk = base + (size_t)idx * DM;
#pragma unroll
      for (int e = 0; e < 16; ++e) d += xreg[e] * xk[lane + 32 * e];
    }
#pragma unroll
    for (int off = 16; off; off >>= 1) d += __shfl_xor(d, off, 32);
    s[l] = (idx >= 0) ? d * 0.04419417382415922f : 0.f;  // zero-pad => score 0
  }
  float mx = s[0];
  for (int l = 1; l < CTX; ++l) mx = fmaxf(mx, s[l]);
  float sum = 0.f;
  for (int l = 0; l < CTX; ++l) { s[l] = expf(s[l] - mx); sum += s[l]; }
  float invs = 1.f / sum;

  float o[16] = {};
  for (int l = 0; l < CTX; ++l) {
    int idx = t - (CTX - 1) + l;
    if (idx < 0) continue;
    float w = s[l] * invs;
    const float* xk = base + (size_t)idx * DM;
#pragma unroll
    for (int e = 0; e < 16; ++e) o[e] += w * xk[lane + 32 * e];
  }
  float* op = Out + (size_t)token * DM;
#pragma unroll
  for (int e = 0; e < 16; ++e) op[lane + 32 * e] = o[e];
}

// ---------------- per-head attention over t (64x64), head slice of DM ----------
__global__ __launch_bounds__(64) void ra_core_attn(
    const float* __restrict__ Q, const float* __restrict__ Kt,
    const float* __restrict__ V, float* __restrict__ Out)
{
  int head = blockIdx.x & (NHEADS - 1);
  int bm   = blockIdx.x >> 3;
  int qi   = threadIdx.x;
  const size_t sb = (size_t)bm * TLEN * DM + head * DK;

  float qreg[DK];
  const float* qrow = Q + sb + (size_t)qi * DM;
#pragma unroll
  for (int d = 0; d < DK; ++d) qreg[d] = qrow[d];

  float s[TLEN];
  float mx = -3.4e38f;
  for (int kj = 0; kj < TLEN; ++kj) {
    const float* krow = Kt + sb + (size_t)kj * DM;
    float d = 0.f;
#pragma unroll
    for (int dd = 0; dd < DK; ++dd) d += qreg[dd] * krow[dd];
    d *= 0.125f;                       // 1/sqrt(64)
    s[kj] = d; mx = fmaxf(mx, d);
  }
  float sum = 0.f;
  for (int kj = 0; kj < TLEN; ++kj) { s[kj] = expf(s[kj] - mx); sum += s[kj]; }
  float invs = 1.f / sum;

  float o[DK] = {};
  for (int kj = 0; kj < TLEN; ++kj) {
    float w = s[kj] * invs;
    const float* vrow = V + sb + (size_t)kj * DM;
#pragma unroll
    for (int d = 0; d < DK; ++d) o[d] += w * vrow[d];
  }
  float* orow = Out + sb + (size_t)qi * DM;
#pragma unroll
  for (int d = 0; d < DK; ++d) orow[d] = o[d];
}

// ---------------- relational attention across m (32 assets) --------------------
// A layout (b,m,t,h,d) flat as (b,m,t,DM). Z layout (m,b,t,h,d) flat.
__global__ __launch_bounds__(32) void ra_rel_attn(
    const float* __restrict__ A, float* __restrict__ Z)
{
  int head = blockIdx.x & (NHEADS - 1);
  int bt   = blockIdx.x >> 3;
  int t    = bt & (TLEN - 1);
  int b    = bt >> 6;
  int mm   = threadIdx.x;

  const size_t hb = (size_t)head * DK;
  auto aoff = [&](int n) {
    return (size_t)((b * MDIM + n) * TLEN + t) * DM + hb;
  };

  float y[DK];
  const float* my = A + aoff(mm);
#pragma unroll
  for (int d = 0; d < DK; ++d) y[d] = my[d];

  float s[MDIM];
  float mx = -3.4e38f;
  for (int n = 0; n < MDIM; ++n) {
    const float* yn = A + aoff(n);
    float d = 0.f;
#pragma unroll
    for (int dd = 0; dd < DK; ++dd) d += y[dd] * yn[dd];
    d *= 0.125f;
    s[n] = d; mx = fmaxf(mx, d);
  }
  float sum = 0.f;
  for (int n = 0; n < MDIM; ++n) { s[n] = expf(s[n] - mx); sum += s[n]; }
  float invs = 1.f / sum;

  float o[DK] = {};
  for (int n = 0; n < MDIM; ++n) {
    float w = s[n] * invs;
    const float* yn = A + aoff(n);
#pragma unroll
    for (int d = 0; d < DK; ++d) o[d] += w * yn[d];
  }
  float* zp = Z + (size_t)((mm * BATCH + b) * TLEN + t) * DM + hb;
#pragma unroll
  for (int d = 0; d < DK; ++d) zp[d] = o[d];
}

// ---------------- decision head ----------------
__global__ __launch_bounds__(32) void ra_decision_dot(
    const float* __restrict__ YN, const float* __restrict__ previous,
    const float* __restrict__ W, const float* __restrict__ bias,
    float* __restrict__ outA)
{
  int bm = blockIdx.x;                 // b*32 + mm
  int b = bm >> 5, mm = bm & 31;
  int lane = threadIdx.x;
  const float* feat = YN + ((size_t)bm * TLEN + (TLEN - 1)) * DM; // last token
  float p = 0.f;
  for (int c = lane; c < DM; c += 32) p += feat[c] * W[c];
#pragma unroll
  for (int off = 16; off; off >>= 1) p += __shfl_xor(p, off, 32);
  if (lane == 0)
    outA[bm] = p + previous[b * (MDIM + 1) + mm + 1] * W[DM] + bias[0];
}

__global__ void ra_decision_final(
    const float* __restrict__ decA, const float* __restrict__ m0,
    const float* __restrict__ m1, const float* __restrict__ m2,
    float* __restrict__ out)
{
  int b = threadIdx.x;
  if (b >= BATCH) return;
  const float sign[3] = {1.f, -1.f, 1.f};
  const float* moneys[3] = {m0, m1, m2};
  float res[MDIM + 1];
  for (int j = 0; j <= MDIM; ++j) res[j] = 0.f;
  for (int hd = 0; hd < 3; ++hd) {
    float v[MDIM + 1];
    v[0] = moneys[hd][0];
    for (int mm = 0; mm < MDIM; ++mm) v[1 + mm] = decA[hd * 64 + b * MDIM + mm];
    float mx = v[0];
    for (int j = 1; j <= MDIM; ++j) mx = fmaxf(mx, v[j]);
    float sum = 0.f;
    for (int j = 0; j <= MDIM; ++j) { v[j] = expf(v[j] - mx); sum += v[j]; }
    float inv = 1.f / sum;
    for (int j = 0; j <= MDIM; ++j) res[j] += sign[hd] * v[j] * inv;
  }
  for (int j = 0; j <= MDIM; ++j) out[b * (MDIM + 1) + j] = res[j];
}

// ===========================================================================
// Host side
// ===========================================================================
struct AttnP { const float *Wq,*bq,*Wk,*bk,*Wv,*bv,*Wo,*bo; };
struct FfnP  { const float *W1,*b1,*W2,*b2; };
struct LNP   { const float *g,*b; };

extern "C" void kernel_launch(void* const* d_in, const int* in_sizes, int n_in,
                              void* d_out, int out_size, void* d_ws, size_t ws_size,
                              hipStream_t stream)
{
  (void)in_sizes; (void)n_in; (void)out_size; (void)ws_size;
  const float* enc_input = (const float*)d_in[0];
  const float* dec_input = (const float*)d_in[1];
  const float* previous  = (const float*)d_in[2];

  int p = 3;
  auto F = [&]() { return (const float*)d_in[p++]; };
  auto getAttn = [&]() { AttnP a;
    a.Wq=F(); a.bq=F(); a.Wk=F(); a.bk=F(); a.Wv=F(); a.bv=F(); a.Wo=F(); a.bo=F();
    return a; };
  auto getFfn = [&]() { FfnP f; f.W1=F(); f.b1=F(); f.W2=F(); f.b2=F(); return f; };
  auto getLN  = [&]() { LNP l; l.g=F(); l.b=F(); return l; };

  const float* embWe = F(); const float* embBe = F();
  const float* embWd = F(); const float* embBd = F();

  AttnP encA[2]; FfnP encF[2]; LNP encL0[2], encL1[2];
  for (int i = 0; i < 2; ++i) {
    encA[i]=getAttn(); encF[i]=getFfn(); encL0[i]=getLN(); encL1[i]=getLN();
  }
  LNP encNorm = getLN();

  AttnP decSelf[2], decSrc[2]; FfnP decF[2]; LNP decL0[2], decL1[2], decL2[2];
  for (int i = 0; i < 2; ++i) {
    decSelf[i]=getAttn(); decSrc[i]=getAttn(); decF[i]=getFfn();
    decL0[i]=getLN(); decL1[i]=getLN(); decL2[i]=getLN();
  }
  LNP decNorm = getLN();

  const float* piW=F(); const float* piB=F();
  const float* psW=F(); const float* psB=F();
  const float* prW=F(); const float* prB=F();
  const float* money=F(); const float* money2=F(); const float* money3=F();

  // ---- workspace slots (each SLOT floats = 8 MB) ----
  float* ws   = (float*)d_ws;
  float* ACT  = ws + 0 * SLOT;   // x / y activations
  float* XN   = ws + 1 * SLOT;   // layernorm output
  float* Qb   = ws + 2 * SLOT;
  float* Kb   = ws + 3 * SLOT;
  float* Vb   = ws + 4 * SLOT;
  float* CQ   = ws + 5 * SLOT;
  float* CK   = ws + 6 * SLOT;
  float* MEM  = ws + 7 * SLOT;   // encoder memory
  float* decA = ws + 8 * SLOT;   // 192 floats (padded to 1024)
  float* ATT  = Qb;              // reuse: att (b,m,t,DM)
  float* Zb   = Kb;              // reuse: rel output (m,b,t,DM) flat
  float* HID  = Qb;              // reuse: FFN hidden 4096x2048 (slots 2..5)
  // bf16 staging: XBF up to 4096x2048 bf16 (16 MB), WTB up to 2048x512 (2 MB)
  bf16_t* XBF = (bf16_t*)(ws + 8 * SLOT + 1024);
  bf16_t* WTB = (bf16_t*)(ws + 8 * SLOT + 1024 + (size_t)4 * 1024 * 1024);

  auto gemm = [&](const float* X, const float* W, const float* bias,
                  const float* res, float* Y, int K, int O, int relu) {
    int nx = NTOK * K, nw = K * O;
    ra_cvt_bf16  <<<(nx + 255) / 256, 256, 0, stream>>>(X, XBF, nx);
    ra_cvt_bf16_T<<<(nw + 255) / 256, 256, 0, stream>>>(W, WTB, K, O);
    dim3 grid(O / 64, NTOK / 128), block(128);
    ra_gemm_wmma<<<grid, block, 0, stream>>>(XBF, WTB, bias, res, Y, K, O, relu);
  };
  auto lnorm = [&](const float* X, const LNP& l, float* Y) {
    ra_layernorm<<<NTOK, 32, 0, stream>>>(X, l.g, l.b, Y);
  };
  auto run_attn = [&](const AttnP& at, const float* qin, const float* kvin, int relat) {
    gemm(qin,  at.Wq, at.bq, nullptr, Qb, DM, DM, 0);
    gemm(kvin, at.Wk, at.bk, nullptr, Kb, DM, DM, 0);
    gemm(kvin, at.Wv, at.bv, nullptr, Vb, DM, DM, 0);
    ra_ctx_attn<<<NTOK, 32, 0, stream>>>(Qb, CQ);
    ra_ctx_attn<<<NTOK, 32, 0, stream>>>(Kb, CK);
    ra_core_attn<<<BATCH * MDIM * NHEADS, TLEN, 0, stream>>>(CQ, CK, Vb, ATT);
    const float* oin = ATT;
    if (relat) {
      ra_rel_attn<<<BATCH * TLEN * NHEADS, MDIM, 0, stream>>>(ATT, Zb);
      oin = Zb;   // (m,b,t,h,d) flat == reference's transpose+reshape
    }
    gemm(oin, at.Wo, at.bo, ACT, ACT, DM, DM, 0);   // residual add fused
  };
  auto run_ffn = [&](const FfnP& f) {
    gemm(XN, f.W1, f.b1, nullptr, HID, DM, 4 * DM, 1);   // ReLU
    gemm(HID, f.W2, f.b2, ACT, ACT, 4 * DM, DM, 0);      // + residual
  };
  auto embed = [&](const float* inp, const float* W, const float* b) {
    gemm(inp, W, b, nullptr, ACT, 32, DM, 0);
    ra_add_pe<<<(NTOK * DM + 255) / 256, 256, 0, stream>>>(ACT);
  };

  // ---------------- encoder ----------------
  embed(enc_input, embWe, embBe);
  for (int i = 0; i < 2; ++i) {
    lnorm(ACT, encL0[i], XN);
    run_attn(encA[i], XN, XN, 1);
    lnorm(ACT, encL1[i], XN);
    run_ffn(encF[i]);
  }
  lnorm(ACT, encNorm, MEM);

  // ---------------- decoder ----------------
  embed(dec_input, embWd, embBd);
  for (int i = 0; i < 2; ++i) {
    lnorm(ACT, decL0[i], XN);
    run_attn(decSelf[i], XN, XN, 1);
    lnorm(ACT, decL1[i], XN);
    run_attn(decSrc[i], XN, MEM, 0);
    lnorm(ACT, decL2[i], XN);
    run_ffn(decF[i]);
  }
  lnorm(ACT, decNorm, XN);   // final normed y in XN

  // ---------------- decision ----------------
  ra_decision_dot<<<BATCH * MDIM, 32, 0, stream>>>(XN, previous, piW, piB, decA + 0);
  ra_decision_dot<<<BATCH * MDIM, 32, 0, stream>>>(XN, previous, psW, psB, decA + 64);
  ra_decision_dot<<<BATCH * MDIM, 32, 0, stream>>>(XN, previous, prW, prB, decA + 128);
  ra_decision_final<<<1, 32, 0, stream>>>(decA, money, money2, money3, (float*)d_out);
}
